// DiffSSM_56770877718745
// MI455X (gfx1250) — matrix operand order repaired and checked by reading, not compiled, not run on hardware
//
#include <hip/hip_runtime.h>
#include <hip/hip_bf16.h>
#include <math.h>

// ---------------- problem constants ----------------
#define B_SZ 8
#define L_SZ 2048
#define D_SZ 1024
#define N_ST 64
#define M_SZ (B_SZ * L_SZ)   // 16384 rows
#define C_CH 16              // scan chunks per sequence
#define T_CH (L_SZ / C_CH)   // 128 steps per chunk

// ---------------- vector types ----------------
typedef __attribute__((ext_vector_type(8)))  __bf16 v8bf;
typedef __attribute__((ext_vector_type(16))) __bf16 v16bf;
typedef __attribute__((ext_vector_type(8)))  float  v8f;
typedef int v4i_t __attribute__((vector_size(16)));   // matches async-LDS builtin param

// CDNA5 async global->LDS copy availability
#if __has_builtin(__builtin_amdgcn_global_load_async_to_lds_b128) && \
    __has_builtin(__builtin_amdgcn_s_wait_asynccnt)
#define USE_ASYNC_COPY 1
#else
#define USE_ASYNC_COPY 0
#endif

__device__ __forceinline__ __bf16 f2bf(float f) {
    unsigned u = __builtin_bit_cast(unsigned, f);
    u += 0x7fffu + ((u >> 16) & 1u);          // round-to-nearest-even
    unsigned short h = (unsigned short)(u >> 16);
    return __builtin_bit_cast(__bf16, h);
}

// packed pair conversion: prefer hardware v_cvt_pk_bf16_f32
__device__ __forceinline__ unsigned pkbf16(float a, float b) {
#if __has_builtin(__builtin_amdgcn_cvt_pk_bf16_f32)
    typedef __attribute__((ext_vector_type(2))) __bf16 v2bf_t;
    v2bf_t r = __builtin_amdgcn_cvt_pk_bf16_f32(a, b);
    return __builtin_bit_cast(unsigned, r);
#else
    unsigned lo = (unsigned)__builtin_bit_cast(unsigned short, f2bf(a));
    unsigned hi = (unsigned)__builtin_bit_cast(unsigned short, f2bf(b));
    return lo | (hi << 16);
#endif
}

// ---------------- WMMA GEMM (bf16 in, f32 acc) ----------------
// MODE 0: C[m,n] = sum_k A[m,k] * W[k,n]            (W row-major KxN)
// MODE 1: C[m,n] = sum_j sum_k A[shift(m,j),k] * W[n,k,j]  (conv1d, pad=1, W is (O,I,3))
// EPI  0: out f32 = acc + bias[n]
// EPI  1: out bf16 = silu(acc + bias[n])
// EPI  2: out bf16 = acc + bias[n] + hf*(b1*ns) + hb*(b2*ns)
#define BM 128
#define BN 128
#define BK 32
#define LDT 40   // padded LDS row stride (elements)

template <int MODE, int EPI, bool A_F32>
__global__ __launch_bounds__(256, 2)
void gemm_wmma_kernel(const void* __restrict__ Aptr,
                      const float* __restrict__ W,
                      const float* __restrict__ bias,
                      void* __restrict__ Out,
                      const float* __restrict__ hf,
                      const float* __restrict__ hb,
                      const float* __restrict__ ns,
                      const float* __restrict__ beta1p,
                      const float* __restrict__ beta2p)
{
    constexpr int Kdim = D_SZ, Ndim = D_SZ;
    __shared__ __align__(16) __bf16 As[BM * LDT];
    __shared__ __align__(16) __bf16 Bs[BN * LDT];

    const int tid = threadIdx.x;
    const int wid = tid >> 5;
    const int lid = tid & 31;
    const int hi  = lid >> 4;      // half-wave select
    const int l16 = lid & 15;
    const int m0  = blockIdx.y * BM;
    const int n0  = blockIdx.x * BN;
    const int wm  = (wid >> 2) * 64;   // wave row offset (0,64)
    const int wn  = (wid & 3) * 32;    // wave col offset (0..96)

    v8f acc[4][2] = {};

    const int nj = (MODE == 1) ? 3 : 1;
    for (int j = 0; j < nj; ++j) {
        for (int kt = 0; kt < Kdim / BK; ++kt) {
            const int kb = kt * BK;
            // ---- load A tile ----
            if (A_F32) {
                const float* A = (const float*)Aptr;
                #pragma unroll
                for (int i = 0; i < 4; ++i) {
                    int c = i * 256 + tid;
                    int row = c >> 3, k4 = (c & 7) * 4;
                    int g = m0 + row;
                    bool valid = true; int src = g;
                    if (MODE == 1) {
                        int l  = g & (L_SZ - 1);
                        int ls = l + j - 1;
                        valid  = (ls >= 0) && (ls < L_SZ);
                        src    = g + j - 1;
                    }
                    float4 v = make_float4(0.f, 0.f, 0.f, 0.f);
                    if (valid) {
                        const float* p = A + (size_t)src * Kdim + kb + k4;
                        v = *(const float4*)p;
                        if (kb + BK < Kdim) __builtin_prefetch(p + BK, 0, 1);
                    }
                    uint2 pk;
                    pk.x = pkbf16(v.x, v.y);
                    pk.y = pkbf16(v.z, v.w);
                    *(uint2*)&As[row * LDT + k4] = pk;
                }
            } else {
                const __bf16* A = (const __bf16*)Aptr;
                #pragma unroll
                for (int i = 0; i < 2; ++i) {
                    int c = i * 256 + tid;
                    int row = c >> 2, k8 = (c & 3) * 8;
                    int g = m0 + row;
                    bool valid = true; int src = g;
                    if (MODE == 1) {
                        int l  = g & (L_SZ - 1);
                        int ls = l + j - 1;
                        valid  = (ls >= 0) && (ls < L_SZ);
                        src    = g + j - 1;
                    }
                    __bf16* dst = &As[row * LDT + k8];
                    const __bf16* sp = A + (size_t)src * Kdim + kb + k8;
#if USE_ASYNC_COPY
                    if (valid) {
                        __builtin_amdgcn_global_load_async_to_lds_b128(
                            (__attribute__((address_space(1))) v4i_t*)sp,
                            (__attribute__((address_space(3))) v4i_t*)dst, 0, 0);
                        if (kb + BK < Kdim) __builtin_prefetch(sp + BK, 0, 1);
                    } else {
                        *(uint4*)dst = make_uint4(0u, 0u, 0u, 0u);
                    }
#else
                    uint4 v = make_uint4(0u, 0u, 0u, 0u);
                    if (valid) {
                        v = *(const uint4*)sp;
                        if (kb + BK < Kdim) __builtin_prefetch(sp + BK, 0, 1);
                    }
                    *(uint4*)dst = v;
#endif
                }
            }
            // ---- load B tile, N-major in LDS: Bs[n][k] ----
            if (MODE == 0) {
                #pragma unroll
                for (int i = 0; i < 4; ++i) {
                    int c = i * 256 + tid;
                    int kk = c >> 5, n4 = (c & 31) * 4;
                    float4 v = *(const float4*)(W + (size_t)(kb + kk) * Ndim + n0 + n4);
                    Bs[(n4 + 0) * LDT + kk] = f2bf(v.x);
                    Bs[(n4 + 1) * LDT + kk] = f2bf(v.y);
                    Bs[(n4 + 2) * LDT + kk] = f2bf(v.z);
                    Bs[(n4 + 3) * LDT + kk] = f2bf(v.w);
                }
            } else {
                #pragma unroll
                for (int i = 0; i < 16; ++i) {
                    int c = i * 256 + tid;
                    int kk = c & 31, nn = c >> 5;
                    float v = W[(size_t)(n0 + nn) * (Kdim * 3) + (size_t)(kb + kk) * 3 + j];
                    Bs[nn * LDT + kk] = f2bf(v);
                }
            }
#if USE_ASYNC_COPY
            if (!A_F32) __builtin_amdgcn_s_wait_asynccnt(0);
#endif
            __syncthreads();

            // ---- build fragments per ISA 16-bit A/B wave32 layouts ----
            v16bf af[4], bfr[2];
            #pragma unroll
            for (int mi = 0; mi < 4; ++mi) {
                int m = wm + mi * 16 + l16;
                int ko = hi ? 8 : 0;                 // lanes>=16 hold K 8-15 / 24-31
                v8bf lo = *(const v8bf*)&As[m * LDT + ko];
                v8bf hs = *(const v8bf*)&As[m * LDT + ko + 16];
                af[mi] = __builtin_shufflevector(lo, hs, 0,1,2,3,4,5,6,7,8,9,10,11,12,13,14,15);
            }
            #pragma unroll
            for (int ni = 0; ni < 2; ++ni) {
                int n = wn + ni * 16 + l16;
                int ko = hi ? 8 : 0;                 // lanes>=16 hold K 8-15 / 24-31
                v8bf lo = *(const v8bf*)&Bs[n * LDT + ko];
                v8bf hs = *(const v8bf*)&Bs[n * LDT + ko + 16];
                bfr[ni] = __builtin_shufflevector(lo, hs, 0,1,2,3,4,5,6,7,8,9,10,11,12,13,14,15);
            }
            #pragma unroll
            for (int mi = 0; mi < 4; ++mi)
                #pragma unroll
                for (int ni = 0; ni < 2; ++ni)
                    acc[mi][ni] = __builtin_amdgcn_wmma_f32_16x16x32_bf16(
                        false, af[mi], false, bfr[ni], (short)0, acc[mi][ni], false, false);
            __syncthreads();
        }
    }

    // ---- epilogue: C/D layout = VGPR e, lanes0-15: M=e,N=lane; lanes16-31: M=e+8 ----
    float b1v = 0.f, b2v = 0.f;
    if (EPI == 2) { b1v = beta1p[0]; b2v = beta2p[0]; }
    #pragma unroll
    for (int mi = 0; mi < 4; ++mi) {
        #pragma unroll
        for (int ni = 0; ni < 2; ++ni) {
            const int nloc = n0 + wn + ni * 16 + l16;
            const float bv = bias[nloc];
            #pragma unroll
            for (int e = 0; e < 8; ++e) {
                const int mloc = m0 + wm + mi * 16 + hi * 8 + e;
                float v = acc[mi][ni][e] + bv;
                const size_t idx = (size_t)mloc * Ndim + nloc;
                if (EPI == 0) {
                    ((float*)Out)[idx] = v;
                } else if (EPI == 1) {
                    float sv = v / (1.f + expf(-v));     // silu
                    ((__bf16*)Out)[idx] = f2bf(sv);
                } else {
                    const int bidx = mloc >> 11;          // row / L
                    const float nsv = ns[(size_t)bidx * Ndim + nloc];
                    v += hf[idx] * (b1v * nsv) + hb[idx] * (b2v * nsv);
                    ((__bf16*)Out)[idx] = f2bf(v);
                }
            }
        }
    }
}

// ---------------- LayerNorm over rows of D=1024 ----------------
template <bool RESID, bool WRITE_BF>
__global__ __launch_bounds__(256)
void ln_kernel(const float* __restrict__ in, const float* __restrict__ g,
               const float* __restrict__ bb, const float* __restrict__ resid,
               float* __restrict__ of, __bf16* __restrict__ obf)
{
    __shared__ float rs[256], rss[256];
    const int row = blockIdx.x;
    const int tid = threadIdx.x;
    const float* rp = in + (size_t)row * D_SZ;
    float v[4], s = 0.f, ss = 0.f;
    #pragma unroll
    for (int e = 0; e < 4; ++e) {
        v[e] = rp[tid + e * 256];
        s += v[e]; ss += v[e] * v[e];
    }
    rs[tid] = s; rss[tid] = ss;
    __syncthreads();
    for (int off = 128; off > 0; off >>= 1) {
        if (tid < off) { rs[tid] += rs[tid + off]; rss[tid] += rss[tid + off]; }
        __syncthreads();
    }
    const float mean = rs[0] * (1.f / D_SZ);
    const float var  = rss[0] * (1.f / D_SZ) - mean * mean;
    const float rstd = rsqrtf(var + 1e-5f);
    #pragma unroll
    for (int e = 0; e < 4; ++e) {
        const int i = tid + e * 256;
        float y = (v[e] - mean) * rstd * g[i] + bb[i];
        if (RESID) y += resid[(size_t)row * D_SZ + i];
        of[(size_t)row * D_SZ + i] = y;
        if (WRITE_BF) obf[(size_t)row * D_SZ + i] = f2bf(y);
    }
}

// ---------------- chunk-parallel diagonal SSM scan ----------------
// y[t] = sum_n s_n * z_n[t] + D0 * cumsum(h),  z_n[t] = lam_n * z_n[t-1] + h[t]
// Pass 1: per (b,d,chunk) local end-state (zero init) + local sum.
// Pass 2a: per (b,d,n) prefix-combine chunk states with lam^T (in place -> init states).
// Pass 2b: per (b,d) exclusive prefix of chunk sums.
// Pass 3: per (b,d,chunk) rescan from true init state, write y.

template <int DIR>
__global__ __launch_bounds__(256)
void scan_p1(const float* __restrict__ h, const float* __restrict__ Amat,
             float* __restrict__ S, float* __restrict__ cumS)
{
    const int gid = blockIdx.x * 256 + threadIdx.x;   // B*C*D threads
    const int d = gid & (D_SZ - 1);
    const int c = (gid >> 10) & (C_CH - 1);
    const int b = gid >> 14;
    float lam[N_ST], z[N_ST];
    #pragma unroll
    for (int n = 0; n < N_ST; ++n) { lam[n] = expf(Amat[n * N_ST + n]); z[n] = 0.f; }
    float cum = 0.f;
    const size_t base = (size_t)b * L_SZ * D_SZ + d;
    for (int t = 0; t < T_CH; ++t) {
        const int i = c * T_CH + t;
        const int pos = (DIR > 0) ? i : (L_SZ - 1 - i);
        const float hv = h[base + (size_t)pos * D_SZ];
        cum += hv;
        #pragma unroll
        for (int n = 0; n < N_ST; ++n) z[n] = fmaf(lam[n], z[n], hv);
    }
    const size_t sb = (size_t)(b * C_CH + c) * N_ST * D_SZ + d;
    #pragma unroll
    for (int n = 0; n < N_ST; ++n) S[sb + (size_t)n * D_SZ] = z[n];
    cumS[(size_t)(b * C_CH + c) * D_SZ + d] = cum;
}

__global__ __launch_bounds__(256)
void scan_p2a(const float* __restrict__ Amat, float* __restrict__ S)
{
    const int gid = blockIdx.x * 256 + threadIdx.x;   // B*N*D threads
    const int d = gid & (D_SZ - 1);
    const int n = (gid >> 10) & (N_ST - 1);
    const int b = gid >> 16;
    const float lamT = expf(Amat[n * N_ST + n] * (float)T_CH);
    float zrun = 0.f;
    for (int c = 0; c < C_CH; ++c) {
        const size_t idx = ((size_t)(b * C_CH + c) * N_ST + n) * D_SZ + d;
        const float f = S[idx];
        S[idx] = zrun;                       // exclusive prefix: init state of chunk c
        zrun = fmaf(lamT, zrun, f);
    }
}

__global__ __launch_bounds__(256)
void scan_p2b(float* __restrict__ cumS)
{
    const int gid = blockIdx.x * 256 + threadIdx.x;   // B*D threads
    const int d = gid & (D_SZ - 1);
    const int b = gid >> 10;
    float run = 0.f;
    for (int c = 0; c < C_CH; ++c) {
        const size_t idx = (size_t)(b * C_CH + c) * D_SZ + d;
        const float s = cumS[idx];
        cumS[idx] = run;
        run += s;
    }
}

template <int DIR>
__global__ __launch_bounds__(256)
void scan_p3(const float* __restrict__ h, const float* __restrict__ Amat,
             const float* __restrict__ Bv, const float* __restrict__ Cv,
             const float* __restrict__ Dv, const float* __restrict__ S,
             const float* __restrict__ cumS, float* __restrict__ out)
{
    const int gid = blockIdx.x * 256 + threadIdx.x;   // B*C*D threads
    const int d = gid & (D_SZ - 1);
    const int c = (gid >> 10) & (C_CH - 1);
    const int b = gid >> 14;
    float lam[N_ST], sw[N_ST], z[N_ST];
    const size_t sb = (size_t)(b * C_CH + c) * N_ST * D_SZ + d;
    #pragma unroll
    for (int n = 0; n < N_ST; ++n) {
        lam[n] = expf(Amat[n * N_ST + n]);
        sw[n]  = Bv[n] * Cv[n];
        z[n]   = S[sb + (size_t)n * D_SZ];
    }
    const float D0 = Dv[0];
    float cum = cumS[(size_t)(b * C_CH + c) * D_SZ + d];
    const size_t base = (size_t)b * L_SZ * D_SZ + d;
    for (int t = 0; t < T_CH; ++t) {
        const int i = c * T_CH + t;
        const int pos = (DIR > 0) ? i : (L_SZ - 1 - i);
        const size_t idx = base + (size_t)pos * D_SZ;
        const float hv = h[idx];
        cum += hv;
        float y = D0 * cum;
        #pragma unroll
        for (int n = 0; n < N_ST; ++n) {
            z[n] = fmaf(lam[n], z[n], hv);
            y    = fmaf(sw[n], z[n], y);
        }
        out[idx] = y;
    }
}

// ---------------- sigmoid(timestep_embedding(t, D)) ----------------
__global__ __launch_bounds__(256)
void ns_kernel(const float* __restrict__ t, float* __restrict__ ns)
{
    const int gid = blockIdx.x * 256 + threadIdx.x;
    if (gid >= B_SZ * D_SZ) return;
    const int b = gid >> 10, i = gid & (D_SZ - 1);
    const int half = D_SZ / 2;
    const int fi = (i < half) ? i : (i - half);
    const float lg = 9.210340371976184f / (float)(half - 1);  // log(1e4)/(half-1)
    const float freq = expf(-lg * (float)fi);
    const float ang = t[b] * freq;
    const float v = (i < half) ? sinf(ang) : cosf(ang);
    ns[gid] = 1.f / (1.f + expf(-v));
}

// ---------------- launch ----------------
extern "C" void kernel_launch(void* const* d_in, const int* in_sizes, int n_in,
                              void* d_out, int out_size, void* d_ws, size_t ws_size,
                              hipStream_t stream)
{
    const float* x   = (const float*)d_in[0];
    const float* t   = (const float*)d_in[1];
    const float* be1 = (const float*)d_in[2];
    const float* be2 = (const float*)d_in[3];
    const float* Wi  = (const float*)d_in[4];
    const float* bi  = (const float*)d_in[5];
    const float* Wo  = (const float*)d_in[6];
    const float* bo  = (const float*)d_in[7];
    const float* w1  = (const float*)d_in[8];
    const float* bc1 = (const float*)d_in[9];
    const float* w2  = (const float*)d_in[10];
    const float* bc2 = (const float*)d_in[11];
    const float* g1  = (const float*)d_in[12];
    const float* b1  = (const float*)d_in[13];
    const float* g2  = (const float*)d_in[14];
    const float* b2  = (const float*)d_in[15];
    const float* Af  = (const float*)d_in[16];
    const float* Bf  = (const float*)d_in[17];
    const float* Cf  = (const float*)d_in[18];
    const float* Df  = (const float*)d_in[19];
    const float* Ab  = (const float*)d_in[20];
    const float* Bb  = (const float*)d_in[21];
    const float* Cb  = (const float*)d_in[22];
    const float* Db  = (const float*)d_in[23];
    float* out = (float*)d_out;

    const size_t MD4 = (size_t)M_SZ * D_SZ * sizeof(float);   // 64 MB
    const size_t MD2 = (size_t)M_SZ * D_SZ * sizeof(__bf16);  // 32 MB
    char* w = (char*)d_ws;
    float*  tmp  = (float*)w;            w += MD4;   // GEMM scratch / scan-state (reused)
    float*  h32  = (float*)w;            w += MD4;   // LN'd h, f32
    float*  hf   = (float*)w;            w += MD4;   // forward scan out
    float*  hb   = (float*)w;            w += MD4;   // backward scan out
    __bf16* hbf  = (__bf16*)w;           w += MD2;   // LN'd h, bf16
    __bf16* c1   = (__bf16*)w;           w += MD2;   // conv1 out (silu'd)
    __bf16* hmix = (__bf16*)w;           w += MD2;   // mixed, bf16
    float*  nsb  = (float*)w;            w += (size_t)B_SZ * D_SZ * sizeof(float);
    float*  cumS = (float*)w;            w += (size_t)B_SZ * C_CH * D_SZ * sizeof(float);
    float*  S    = tmp;   // 33.5 MB chunk-state buffer, tmp is idle during scans

    dim3 gg(D_SZ / BN, M_SZ / BM), gb(256);
    const int gP1 = (B_SZ * C_CH * D_SZ) / 256;   // 512
    const int gP2a = (B_SZ * N_ST * D_SZ) / 256;  // 2048
    const int gP2b = (B_SZ * D_SZ) / 256;         // 32

    // 1) tmp = x @ Wi + bi
    gemm_wmma_kernel<0, 0, true><<<gg, gb, 0, stream>>>(
        x, Wi, bi, tmp, nullptr, nullptr, nullptr, nullptr, nullptr);
    // 2) h = LN(tmp) -> f32 + bf16   (tmp free after this)
    ln_kernel<false, true><<<M_SZ, 256, 0, stream>>>(tmp, g1, b1, nullptr, h32, hbf);
    // 3) forward scan (chunk-parallel)
    scan_p1<1><<<gP1, 256, 0, stream>>>(h32, Af, S, cumS);
    scan_p2a<<<gP2a, 256, 0, stream>>>(Af, S);
    scan_p2b<<<gP2b, 256, 0, stream>>>(cumS);
    scan_p3<1><<<gP1, 256, 0, stream>>>(h32, Af, Bf, Cf, Df, S, cumS, hf);
    // 4) backward scan
    scan_p1<-1><<<gP1, 256, 0, stream>>>(h32, Ab, S, cumS);
    scan_p2a<<<gP2a, 256, 0, stream>>>(Ab, S);
    scan_p2b<<<gP2b, 256, 0, stream>>>(cumS);
    scan_p3<-1><<<gP1, 256, 0, stream>>>(h32, Ab, Bb, Cb, Db, S, cumS, hb);
    // 5) noise scale
    ns_kernel<<<gP2b, 256, 0, stream>>>(t, nsb);
    // 6) c1 = silu(conv1d(h, w1, bc1))
    gemm_wmma_kernel<1, 1, false><<<gg, gb, 0, stream>>>(
        hbf, w1, bc1, c1, nullptr, nullptr, nullptr, nullptr, nullptr);
    // 7) hmix = conv1d(c1, w2, bc2) + hf*(b1*ns) + hb*(b2*ns)
    gemm_wmma_kernel<1, 2, false><<<gg, gb, 0, stream>>>(
        c1, w2, bc2, hmix, hf, hb, nsb, be1, be2);
    // 8) tmp = hmix @ Wo + bo
    gemm_wmma_kernel<0, 0, false><<<gg, gb, 0, stream>>>(
        hmix, Wo, bo, tmp, nullptr, nullptr, nullptr, nullptr, nullptr);
    // 9) out = x + LN(tmp)
    ln_kernel<true, false><<<M_SZ, 256, 0, stream>>>(tmp, g2, b2, x, out, nullptr);
}
